// EmbeddingHmm_49830210568607
// MI455X (gfx1250) — matrix operand order repaired and checked by reading, not compiled, not run on hardware
//
#include <hip/hip_runtime.h>

#define BSZ     64
#define LSEQ    4096
#define TSTEPS  5
#define TL      84
#define HALO    6
#define TE      96          // TL + 2*HALO, 6 bands of 16
#define NTHR    192         // 6 wave32s
#define PITCH16 66          // u16 pitch for bf16 row tiles (bank-conflict pad)
#define PITCH32 33          // same pitch in u32 units
#define WPITCH  65          // u32 pitch for packed weight tiles
#define NTILES  49          // ceil(4096/84)

typedef __attribute__((ext_vector_type(16))) __bf16 v16bf;
typedef __attribute__((ext_vector_type(8)))  float  v8f;

union V16U { v16bf v; unsigned u[8]; };

__device__ __forceinline__ unsigned short f2bf(float f) {
  unsigned u = __float_as_uint(f);
  unsigned r = ((u >> 16) & 1u) + 0x7FFFu;   // round-to-nearest-even
  return (unsigned short)((u + r) >> 16);
}
__device__ __forceinline__ float bf2f(unsigned short h) {
  return __uint_as_float(((unsigned)h) << 16);
}
__device__ __forceinline__ unsigned pack2(float a, float b) {
  return (unsigned)f2bf(a) | ((unsigned)f2bf(b) << 16);
}

// A-operand: 16x32 bf16 tile. lane: M = lane%16; half = lane/16.
// VGPR j<4 holds K = kb*32 + half*8 + 2j..+1 ; j>=4 holds K = kb*32 + 16 + half*8 + 2(j-4)..+1
__device__ __forceinline__ v16bf load_aop(const unsigned* buf32, int row, int kb, int hh) {
  V16U r;
  int base = row * PITCH32 + kb * 16 + hh * 4;
#pragma unroll
  for (int j = 0; j < 8; ++j) {
    int idx = (j < 4) ? (base + j) : (base + 4 + j);  // j>=4: base + 8 + (j-4)
    r.u[j] = buf32[idx];
  }
  return r.v;
}

// B-operand: 32x16 bf16 tile from pair-packed weights W[k/2][n].
// lane: N = lane%16 (+16*ncol); half = lane/16; VGPR j<4 holds K = kb*32 + half*8 + 2j..+1 ; j>=4 holds K = kb*32 + 16 + half*8 + 2(j-4)..+1
__device__ __forceinline__ v16bf load_bop(const unsigned* w, int ncol, int kb, int lane) {
  V16U r;
  int hh = lane >> 4;
  int n = (lane & 15) + ncol * 16;
#pragma unroll
  for (int j = 0; j < 8; ++j)
    r.u[j] = w[(kb * 16 + hh * 4 + ((j < 4) ? j : (j + 4))) * WPITCH + n];
  return r.v;
}

// S-operand: S[m][k] = F[m-1][k] + B[m+1][k], built as bf16 pairs on the fly
__device__ __forceinline__ v16bf load_sop(const unsigned* fB, const unsigned* bB,
                                          int rf, int rb, int kb, int hh) {
  V16U r;
  int of = rf * PITCH32 + kb * 16 + hh * 4;
  int ob = rb * PITCH32 + kb * 16 + hh * 4;
#pragma unroll
  for (int j = 0; j < 8; ++j) {
    int o = (j < 4) ? j : (j + 4);
    unsigned uf = fB[of + o], ub = bB[ob + o];
    float lo = bf2f((unsigned short)(uf & 0xffffu)) + bf2f((unsigned short)(ub & 0xffffu));
    float hi = bf2f((unsigned short)(uf >> 16))     + bf2f((unsigned short)(ub >> 16));
    r.u[j] = pack2(lo, hi);
  }
  return r.v;
}

__global__ __launch_bounds__(NTHR) void hmm_fused(
    const float* __restrict__ x,  const float* __restrict__ W1, const float* __restrict__ b1,
    const float* __restrict__ W2, const float* __restrict__ b2,
    const float* __restrict__ W3, const float* __restrict__ b3,
    const float* __restrict__ W4, const float* __restrict__ b4,
    float* __restrict__ out) {
  __shared__ unsigned short sA[TE * PITCH16];
  __shared__ unsigned short sF[TE * PITCH16];
  __shared__ unsigned short sB[TE * PITCH16];
  __shared__ unsigned sW2[32 * WPITCH];
  __shared__ unsigned sW4[32 * WPITCH];
  __shared__ float sx[4 * TE];
  __shared__ float sb1[64], sb2[64], sb34[64];
  __shared__ float sW1[64 * 4], sW3[64 * 4];

  const int tid  = threadIdx.x;
  const int lane = tid & 31;
  const int wv   = tid >> 5;
  const int hh   = lane >> 4;
  const int ln   = lane & 15;
  const int m0   = wv * 16;
  const int bidx = blockIdx.y;
  const int gl0  = (int)blockIdx.x * TL - HALO;   // global l of LDS row 0

  // ---- load weights into LDS (bf16 pair-packed B-operand layout) ----
  for (int i = tid; i < 2048; i += NTHR) {
    int e = i & 63, kk = i >> 6;   // W2P[kk][e] = pack(W2[e][2kk], W2[e][2kk+1])
    sW2[kk * WPITCH + e] = pack2(W2[e * 64 + 2 * kk], W2[e * 64 + 2 * kk + 1]);
    sW4[kk * WPITCH + e] = pack2(W4[e * 64 + 2 * kk], W4[e * 64 + 2 * kk + 1]);
  }
  if (tid < 64) {
    sb1[tid]  = b1[tid];
    sb2[tid]  = b2[tid];
    sb34[tid] = b3[tid] + 2.0f * b4[tid];
#pragma unroll
    for (int s = 0; s < 4; ++s) { sW1[tid*4+s] = W1[tid*4+s]; sW3[tid*4+s] = W3[tid*4+s]; }
  }
  for (int i = tid; i < 4 * TE; i += NTHR) {
    int s = i / TE, m = i % TE, l = gl0 + m;
    sx[i] = (l >= 0 && l < LSEQ) ? x[((size_t)bidx * 4 + s) * LSEQ + l] : 0.0f;
  }
  __syncthreads();

  unsigned* A32 = (unsigned*)sA;
  unsigned* F32 = (unsigned*)sF;
  unsigned* B32 = (unsigned*)sB;

  // ---- A = x W1^T + b1 (bf16), F = B = 0. OOB rows get -1e30 so relu pins them to 0. ----
  {
    int row = tid >> 1, c0 = (tid & 1) * 32;
    int l = gl0 + row;
    bool ok = (l >= 0 && l < LSEQ);
    float x0 = sx[row], x1 = sx[TE + row], x2 = sx[2*TE + row], x3 = sx[3*TE + row];
#pragma unroll
    for (int j = 0; j < 16; ++j) {
      int d0 = c0 + 2 * j;
      float a0 = -1e30f, a1 = -1e30f;
      if (ok) {
        a0 = sb1[d0]   + x0*sW1[d0*4]     + x1*sW1[d0*4+1]     + x2*sW1[d0*4+2]     + x3*sW1[d0*4+3];
        a1 = sb1[d0+1] + x0*sW1[(d0+1)*4] + x1*sW1[(d0+1)*4+1] + x2*sW1[(d0+1)*4+2] + x3*sW1[(d0+1)*4+3];
      }
      int w = row * PITCH32 + (d0 >> 1);
      A32[w] = pack2(a0, a1);
      F32[w] = 0u;
      B32[w] = 0u;
    }
  }
  __syncthreads();

  // ---- preload W2 B-operands (live across the T loop) ----
  v16bf w2op[4][2];
#pragma unroll
  for (int nc = 0; nc < 4; ++nc)
#pragma unroll
    for (int kb = 0; kb < 2; ++kb) w2op[nc][kb] = load_bop(sW2, nc, kb, lane);

  // Shifted A-operand row bases: F-chain reads rows [m0-1 .. m0+14], B-chain [m0+1 .. m0+16]
  int rF = m0 - 1 + ln; if (rF < 0) rF = 0;           // clamp only touches halo rows
  int rB = m0 + 1 + ln; if (rB > TE - 1) rB = TE - 1;

  // ---- T recurrence steps, fully in LDS ----
  for (int t = 0; t < TSTEPS; ++t) {
    v16bf aF0 = load_aop(F32, rF, 0, hh);
    v16bf aF1 = load_aop(F32, rF, 1, hh);
    v16bf aB0 = load_aop(B32, rB, 0, hh);
    v16bf aB1 = load_aop(B32, rB, 1, hh);
    v8f gf[4], gb[4];
#pragma unroll
    for (int nc = 0; nc < 4; ++nc) {
      v8f accf = {};
      accf = __builtin_amdgcn_wmma_f32_16x16x32_bf16(false, aF0, false, w2op[nc][0], (short)0, accf, false, false);
      accf = __builtin_amdgcn_wmma_f32_16x16x32_bf16(false, aF1, false, w2op[nc][1], (short)0, accf, false, false);
      gf[nc] = accf;
      v8f accb = {};
      accb = __builtin_amdgcn_wmma_f32_16x16x32_bf16(false, aB0, false, w2op[nc][0], (short)0, accb, false, false);
      accb = __builtin_amdgcn_wmma_f32_16x16x32_bf16(false, aB1, false, w2op[nc][1], (short)0, accb, false, false);
      gb[nc] = accb;
    }
    __syncthreads();   // all waves done reading F/B before anyone overwrites
#pragma unroll
    for (int nc = 0; nc < 4; ++nc) {
      int n = nc * 16 + ln;
      float bb = sb2[n];
#pragma unroll
      for (int r = 0; r < 8; ++r) {
        int m = m0 + r + hh * 8;                      // D-matrix row for this lane/VGPR
        float av = bf2f(sA[m * PITCH16 + n]);
        float fv = av + gf[nc][r] + bb; fv = fv > 0.f ? fv : 0.f;
        float bv = av + gb[nc][r] + bb; bv = bv > 0.f ? bv : 0.f;
        sF[m * PITCH16 + n] = f2bf(fv);
        sB[m * PITCH16 + n] = f2bf(bv);
      }
    }
    __syncthreads();
  }

  // ---- output: miu = relu(x W3^T + b3 + 2*b4 + (F[l-1]+B[l+1]) W4^T) ----
  {
    int rf = m0 + ln - 1; if (rf < 0) rf = 0;
    int rb = m0 + ln + 1; if (rb > TE - 1) rb = TE - 1;
    v16bf s0 = load_sop(F32, B32, rf, rb, 0, hh);
    v16bf s1 = load_sop(F32, B32, rf, rb, 1, hh);
#pragma unroll
    for (int nc = 0; nc < 4; ++nc) {
      v16bf wb0 = load_bop(sW4, nc, 0, lane);
      v16bf wb1 = load_bop(sW4, nc, 1, lane);
      v8f acc = {};
      acc = __builtin_amdgcn_wmma_f32_16x16x32_bf16(false, s0, false, wb0, (short)0, acc, false, false);
      acc = __builtin_amdgcn_wmma_f32_16x16x32_bf16(false, s1, false, wb1, (short)0, acc, false, false);
      int n = nc * 16 + ln;
      float cbase = sb34[n];
      float w30 = sW3[n*4], w31 = sW3[n*4+1], w32 = sW3[n*4+2], w33 = sW3[n*4+3];
#pragma unroll
      for (int r = 0; r < 8; ++r) {
        int m = m0 + r + hh * 8;
        int l = gl0 + m;
        if (m >= HALO && m < HALO + TL && l < LSEQ) {   // interior rows only (l >= 0 implied)
          float c3 = cbase + sx[m]*w30 + sx[TE+m]*w31 + sx[2*TE+m]*w32 + sx[3*TE+m]*w33;
          float v = c3 + acc[r]; v = v > 0.f ? v : 0.f;
          out[((size_t)bidx * LSEQ + l) * 64 + n] = v;
        }
      }
    }
  }
}

extern "C" void kernel_launch(void* const* d_in, const int* in_sizes, int n_in,
                              void* d_out, int out_size, void* d_ws, size_t ws_size,
                              hipStream_t stream) {
  const float* x  = (const float*)d_in[0];
  const float* W1 = (const float*)d_in[1];
  const float* b1 = (const float*)d_in[2];
  const float* W2 = (const float*)d_in[3];
  const float* b2 = (const float*)d_in[4];
  const float* W3 = (const float*)d_in[5];
  const float* b3 = (const float*)d_in[6];
  const float* W4 = (const float*)d_in[7];
  const float* b4 = (const float*)d_in[8];
  // d_in[9] is T (== 5, fixed in the reference; baked in as TSTEPS)
  float* out = (float*)d_out;
  dim3 grid(NTILES, BSZ);
  hmm_fused<<<grid, NTHR, 0, stream>>>(x, W1, b1, W2, b2, W3, b3, W4, b4, out);
}
